// RNNEmbed_85942295593404
// MI455X (gfx1250) — matrix operand order repaired and checked by reading, not compiled, not run on hardware
//
#include <hip/hip_runtime.h>
#include <math.h>

// MI455X / gfx1250 fused Fourier-encode + per-neuron LSTM scan.
// One workgroup per neuron (grid=64), 256 threads = 8 wave32 waves.
// W_hh lives in LDS as f16; W_ih B-fragments cached in registers;
// per timestep each wave does 24 v_wmma_f32_16x16x32_f16.

typedef _Float16 half16 __attribute__((ext_vector_type(16)));
typedef float    float8 __attribute__((ext_vector_type(8)));

#define B_    8
#define T_    1024
#define N_    64
#define H_    128
#define G4_   512    // 4*H
#define INP_  64     // 2*WIDTH
#define WIDTH_ 32

// A-fragment (16x32 f16) load, contiguous K order:
// lane m=l%16; lanes<16: halfs0-15 = K k0..k0+15
// lanes>=16: same with k0 += 16.
__device__ __forceinline__ half16 load_a_frag(const _Float16* base, int stride,
                                              int m, int k0) {
    const _Float16* p = base + m * stride + k0;
    half16 a;
#pragma unroll
    for (int i = 0; i < 16; ++i) { a[i] = p[i]; }
    return a;
}

__device__ __forceinline__ float sigf(float x) { return 1.0f / (1.0f + __expf(-x)); }
__device__ __forceinline__ float tanhfast(float x) {
    return 2.0f / (1.0f + __expf(-2.0f * x)) - 1.0f;
}

__global__ __launch_bounds__(256, 1) void fused_lstm_kernel(
    const float* __restrict__ spikes,   // [B,T,N,1]
    const float* __restrict__ enc,      // [32]
    const float* __restrict__ W_ih,     // [N,512,64]
    const float* __restrict__ W_hh,     // [N,512,128]
    const float* __restrict__ b_ih,     // [N,512]
    const float* __restrict__ b_hh,     // [N,512]
    float* __restrict__ out)            // [B,T,N,H]
{
    __shared__ __align__(32) _Float16 sWhh[G4_ * H_];   // 128 KB
    __shared__ __align__(32) _Float16 sH[16 * H_];      // 4 KB (rows 8..15 zero)
    __shared__ __align__(32) _Float16 sX[16 * INP_];    // 2 KB
    __shared__ float sGates[B_ * G4_];                  // 16 KB
    __shared__ float sEnc[WIDTH_];

    const int n    = blockIdx.x;
    const int tid  = threadIdx.x;
    const int wave = tid >> 5;
    const int lane = tid & 31;
    const int mrow = lane & 15;
    const int koffA = (lane < 16) ? 0 : 16;   // A-frag K sub-offset
    const int koffB = (lane < 16) ? 0 : 16;   // B-frag K sub-offset

    // ---- init: W_hh -> f16 LDS, zero h, encoding weights ----
    const float* whh_g = W_hh + (size_t)n * (G4_ * H_);
    for (int i = tid; i < G4_ * H_; i += 256) sWhh[i] = (_Float16)whh_g[i];
    for (int i = tid; i < 16 * H_; i += 256)  sH[i]   = (_Float16)0.0f;
    if (tid < WIDTH_) sEnc[tid] = enc[tid];

    // ---- cache input-projection B-fragments (time-invariant) + bias ----
    // B-frag layout: lane col = l%16 -> gate row g; lanes<16 K=k..k+15, lanes>=16 K=k+16..k+31
    half16 wihf[8];
    float  bias[4];
    const float* wih_g = W_ih + (size_t)n * (G4_ * INP_);
#pragma unroll
    for (int q = 0; q < 4; ++q) {
        int gcol = (wave * 4 + q) * 16 + mrow;
        bias[q] = b_ih[n * G4_ + gcol] + b_hh[n * G4_ + gcol];
#pragma unroll
        for (int kt = 0; kt < 2; ++kt) {
            const float* wp = wih_g + (size_t)gcol * INP_ + kt * 32 + koffB;
            half16 f;
#pragma unroll
            for (int i = 0; i < 16; ++i) f[i] = (_Float16)wp[i];
            wihf[q * 2 + kt] = f;
        }
    }

    float cst[4] = {0.f, 0.f, 0.f, 0.f};   // persistent cell state (4 elems/thread)

    __syncthreads();

    for (int t = 0; t < T_; ++t) {
        // ---- phase A: fourier-encode x tile [16 x 64] (pad rows 8..15 = 0) ----
#pragma unroll
        for (int j = 0; j < 4; ++j) {
            int e = tid * 4 + j;
            int row = e >> 6, col = e & 63;
            _Float16 v = (_Float16)0.0f;
            if (row < B_) {
                float s  = spikes[((size_t)row * T_ + t) * N_ + n];
                float xv = 2.0f * s - 1.0f;
                float ang = 6.28318530718f * sEnc[col & 31] * xv;
                v = (_Float16)((col < 32) ? __cosf(ang) : __sinf(ang));
            }
            sX[e] = v;
        }
        if (t + 1 < T_) __builtin_prefetch(&spikes[(size_t)(t + 1) * N_ + n], 0, 0);
        __syncthreads();

        // ---- phase B: gates = x@W_ih^T + h@W_hh^T + bias (4 tiles / wave) ----
#pragma unroll
        for (int q = 0; q < 4; ++q) {
            int gbase = (wave * 4 + q) * 16;
            int gcol  = gbase + mrow;
            float8 acc = {};
#pragma unroll
            for (int kt = 0; kt < 2; ++kt) {           // input projection, K=64
                half16 a = load_a_frag(sX, INP_, mrow, kt * 32 + koffA);
                acc = __builtin_amdgcn_wmma_f32_16x16x32_f16(
                        false, a, false, wihf[q * 2 + kt], (short)0, acc, false, false);
            }
#pragma unroll
            for (int kt = 0; kt < 4; ++kt) {           // recurrent, K=128
                half16 a = load_a_frag(sH, H_, mrow, kt * 32 + koffA);
                const half16* bp =
                    (const half16*)&sWhh[(size_t)gcol * H_ + kt * 32 + koffB];
                acc = __builtin_amdgcn_wmma_f32_16x16x32_f16(
                        false, a, false, *bp, (short)0, acc, false, false);
            }
            // D layout: lanes 0..15 hold rows M=0..7 (valid batches) in VGPRs 0..7
            if (lane < 16) {
#pragma unroll
                for (int r = 0; r < 8; ++r)
                    sGates[r * G4_ + gcol] = acc[r] + bias[q];
            }
        }
        __syncthreads();

        // ---- phase C: pointwise LSTM update ----
#pragma unroll
        for (int j = 0; j < 4; ++j) {
            int e  = tid * 4 + j;
            int b  = e >> 7, hx = e & 127;
            float ig = sigf(sGates[b * G4_ + hx]);
            float fg = sigf(sGates[b * G4_ + 128 + hx]);
            float gg = tanhfast(sGates[b * G4_ + 256 + hx]);
            float og = sigf(sGates[b * G4_ + 384 + hx]);
            float c  = fg * cst[j] + ig * gg;
            cst[j]   = c;
            float hv = og * tanhfast(c);
            sH[b * H_ + hx] = (_Float16)hv;
            out[(((size_t)b * T_ + t) * N_ + n) * H_ + hx] = hv;
        }
        __syncthreads();
    }
}

extern "C" void kernel_launch(void* const* d_in, const int* in_sizes, int n_in,
                              void* d_out, int out_size, void* d_ws, size_t ws_size,
                              hipStream_t stream) {
    (void)in_sizes; (void)n_in; (void)d_ws; (void)ws_size; (void)out_size;
    const float* spikes = (const float*)d_in[0];
    const float* enc    = (const float*)d_in[1];
    const float* W_ih   = (const float*)d_in[2];
    const float* W_hh   = (const float*)d_in[3];
    const float* b_ih   = (const float*)d_in[4];
    const float* b_hh   = (const float*)d_in[5];
    float* out = (float*)d_out;

    fused_lstm_kernel<<<dim3(N_), dim3(256), 0, stream>>>(
        spikes, enc, W_ih, W_hh, b_ih, b_hh, out);
}
